// Stem_29386166239247
// MI455X (gfx1250) — matrix operand order repaired and checked by reading, not compiled, not run on hardware
//
#include <hip/hip_runtime.h>
#include <hip/hip_bf16.h>
#include <stdint.h>
#include <math.h>

typedef __attribute__((ext_vector_type(16))) _Float16 v16h;
typedef __attribute__((ext_vector_type(8)))  float    v8f;

#define KNN  9
#define TAPS 9

// ---------------------------------------------------------------------------
// Pre-pack conv kernel (Cout, Cin, 9) f32 into f16 WMMA B-matrix fragments.
// Fragment layout (ISA 7.12.2, 16-bit B 32x16):
//   dword index = ((tile*nchunks + q)*32 + lane)*8 + j
//   holds K = q*32 + (lane>=16 ? 8:0) + 2j + (j>=4 ? 8:0) .. +1,  N = tile*16 + (lane&15)
//   with K linearized as c*9 + t.
// ---------------------------------------------------------------------------
__global__ void build_bfrag_kernel(const float* __restrict__ kern,
                                   uint32_t* __restrict__ frag,
                                   int Cin, int Cout, int nchunks, int ntiles) {
  int i = blockIdx.x * blockDim.x + threadIdx.x;
  int total = ntiles * nchunks * 32 * 8;
  if (i >= total) return;
  int j    = i & 7;
  int l    = (i >> 3) & 31;
  int qt   = i >> 8;
  int q    = qt % nchunks;
  int tile = qt / nchunks;
  int o     = tile * 16 + (l & 15);
  int kbase = q * 32 + ((l >= 16) ? 8 : 0) + 2 * j + ((j >= 4) ? 8 : 0);
  int KD    = Cin * TAPS;
  union { _Float16 h[2]; uint32_t u; } pk;
  pk.h[0] = (_Float16)0.f; pk.h[1] = (_Float16)0.f;
  if (o < Cout) {
    if (kbase < KD) {
      int c = kbase / TAPS, t = kbase - c * TAPS;
      pk.h[0] = (_Float16)kern[(o * Cin + c) * TAPS + t];
    }
    if (kbase + 1 < KD) {
      int c = (kbase + 1) / TAPS, t = (kbase + 1) - c * TAPS;
      pk.h[1] = (_Float16)kern[(o * Cin + c) * TAPS + t];
    }
  }
  frag[i] = pk.u;
}

// ---------------------------------------------------------------------------
// Fused interp-conv (+ optional fused BN/ReLU of the *input* features).
// One wave per output point; grid exactly Nout/4 blocks x 128 threads so all
// control flow is uniform and EXEC is all-ones at every WMMA.
// ---------------------------------------------------------------------------
template <int CIN, int COUT, int NCHUNKS, int NTILES, bool FUSE_BN>
__global__ __launch_bounds__(128) void interp_conv_wmma_kernel(
    const float* __restrict__ feat,    // (B=16, CIN, Nin)  raw (pre-BN) values
    const int*   __restrict__ idx,     // (Nout, 9)
    const float* __restrict__ pos_in,  // (Nin, 2)
    const float* __restrict__ pos_out, // (Nout, 2)
    const float* __restrict__ w1,      // (2,16)
    const float* __restrict__ b1,      // (16)
    const float* __restrict__ w2,      // (16,9)
    const float* __restrict__ b2,      // (9)
    const uint32_t* __restrict__ Bfrag,// packed B fragments
    const float* __restrict__ bias,    // (COUT)
    const float* __restrict__ bn_ss,   // scale[CIN], shift[CIN] for input BN
    float* __restrict__ out,           // (B, COUT, Nout) raw conv output
    int Nin, int Nout)
{
  constexpr int KP = NCHUNKS * 32;
  constexpr int KD = CIN * TAPS;

  __shared__ float    sA[4][96];            // per-wave A[k][t]
  __shared__ _Float16 sTap[4][16 * KP];     // per-wave taps staging, b-major

  const int lane = threadIdx.x & 31;
  const int w    = threadIdx.x >> 5;
  const int n    = blockIdx.x * 4 + w;      // < Nout by grid construction

  // neighbor indices (uniform within wave -> scalar regs)
  int nb[KNN];
  #pragma unroll
  for (int k = 0; k < KNN; ++k) nb[k] = idx[n * KNN + k];
  const float px = pos_out[n * 2 + 0];
  const float py = pos_out[n * 2 + 1];

  // ---- stage 1: weight-net  A[k][t] = (relu(off @ w1 + b1) @ w2 + b2) ----
  for (int p = lane; p < KNN * TAPS; p += 32) {
    int k = p / TAPS, t = p - k * TAPS;
    float ox = pos_in[nb[k] * 2 + 0] - px;
    float oy = pos_in[nb[k] * 2 + 1] - py;
    float a = b2[t];
    #pragma unroll
    for (int jj = 0; jj < 16; ++jj) {
      float hh = ox * w1[jj] + oy * w1[16 + jj] + b1[jj];
      hh = hh > 0.f ? hh : 0.f;
      a += hh * w2[jj * TAPS + t];
    }
    sA[w][p] = a;
  }
  __syncthreads();

  // ---- stage 2: gather (+fused BN/ReLU) and taps contraction -> LDS f16 ----
  for (int p = lane; p < 16 * CIN; p += 32) {
    int b = p & 15, c = p >> 4;
    const float* fr = feat + ((size_t)b * CIN + c) * (size_t)Nin;
    float sc = 1.f, sh = 0.f;
    if (FUSE_BN) { sc = bn_ss[c]; sh = bn_ss[CIN + c]; }
    float fv[KNN];
    #pragma unroll
    for (int k = 0; k < KNN; ++k) {
      float v = fr[nb[k]];
      if (FUSE_BN) { v = v * sc + sh; v = v > 0.f ? v : 0.f; }
      fv[k] = v;
    }
    #pragma unroll
    for (int t = 0; t < TAPS; ++t) {
      float s = 0.f;
      #pragma unroll
      for (int k = 0; k < KNN; ++k) s += fv[k] * sA[w][k * TAPS + t];
      sTap[w][b * KP + c * TAPS + t] = (_Float16)s;
    }
  }
  if (KP > KD) {                       // zero K-padding (layer 1: 27 -> 32)
    constexpr int PAD = KP - KD;
    for (int p = lane; p < 16 * PAD; p += 32) {
      int b = p / PAD, kk = KD + (p - b * PAD);
      sTap[w][b * KP + kk] = (_Float16)0.f;
    }
  }
  __syncthreads();

  // ---- stage 3: out[b,o] = taps(16 x KP) @ B(KP x COUT) via f16 WMMA ----
  v8f acc[NTILES];
  #pragma unroll
  for (int t = 0; t < NTILES; ++t) acc[t] = v8f{};

  const uint4* tq   = (const uint4*)(&sTap[w][0]);
  const int    m    = lane & 15;
  const int    khi  = (lane >= 16) ? 8 : 0;   // A-layout upper-lane K offset (halves)

  for (int q = 0; q < NCHUNKS; ++q) {
    // A fragment: two contiguous 16B runs in the staging buffer
    union { uint4 v4[2]; uint32_t u[8]; v16h h; } av;
    const int db = (m * KP + q * 32 + khi) >> 1;  // dword base, 16B aligned
    av.v4[0] = tq[db >> 2];                       // K 0..7   (lanes>=16: 8..15)
    av.v4[1] = tq[(db >> 2) + 2];                 // K 16..23 (lanes>=16: 24..31)
    #pragma unroll
    for (int tile = 0; tile < NTILES; ++tile) {
      union { uint4 v4[2]; uint32_t u[8]; v16h h; } bv;
      const uint4* bq = (const uint4*)(Bfrag + ((size_t)(tile * NCHUNKS + q) * 32 + lane) * 8);
      bv.v4[0] = bq[0];
      bv.v4[1] = bq[1];
      acc[tile] = __builtin_amdgcn_wmma_f32_16x16x32_f16(false, av.h, false, bv.h,
                                                         (short)0, acc[tile], false, false);
    }
  }

  // ---- epilogue: C/D layout -> out[b][o][n] (+conv bias) ----
  const int nn   = lane & 15;
  const int brow = (lane >= 16) ? 8 : 0;
  float bs[NTILES];
  #pragma unroll
  for (int t = 0; t < NTILES; ++t) bs[t] = bias[t * 16 + nn];
  #pragma unroll
  for (int r = 0; r < 8; ++r) {
    int b = brow + r;
    size_t base = ((size_t)b * COUT) * (size_t)Nout + n;
    #pragma unroll
    for (int t = 0; t < NTILES; ++t)
      out[base + (size_t)(t * 16 + nn) * Nout] = acc[t][r] + bs[t];
  }
}

// ---------------------------------------------------------------------------
// BatchNorm stats on raw conv output: one block per channel -> scale/shift.
// ---------------------------------------------------------------------------
__global__ void bn_stats_kernel(const float* __restrict__ h,
                                const float* __restrict__ g,
                                const float* __restrict__ bb,
                                float* __restrict__ ss,
                                int C, int N, int Btot) {
  __shared__ float s1[256], s2[256];
  int c = blockIdx.x;
  float a = 0.f, q = 0.f;
  int total = Btot * N;
  for (int i = threadIdx.x; i < total; i += blockDim.x) {
    int b = i / N, n = i - b * N;
    float v = h[((size_t)b * C + c) * (size_t)N + n];
    a += v; q += v * v;
  }
  s1[threadIdx.x] = a; s2[threadIdx.x] = q;
  __syncthreads();
  for (int sft = 128; sft > 0; sft >>= 1) {
    if (threadIdx.x < sft) {
      s1[threadIdx.x] += s1[threadIdx.x + sft];
      s2[threadIdx.x] += s2[threadIdx.x + sft];
    }
    __syncthreads();
  }
  if (threadIdx.x == 0) {
    float inv  = 1.f / (float)total;
    float mean = s1[0] * inv;
    float var  = s2[0] * inv - mean * mean;
    float sc   = rsqrtf(var + 1e-5f) * g[c];
    ss[c]      = sc;
    ss[C + c]  = bb[c] - mean * sc;
  }
}

// ---------------------------------------------------------------------------
// Final pool with fused BN/ReLU of layer-3 output:
//   out[b,c,m] = max_j relu(h[b,c,pidx[m,j]] * sc[c] + sh[c])
// ---------------------------------------------------------------------------
__global__ void pool_bn_max_kernel(const float* __restrict__ h,
                                   const int* __restrict__ pidx,
                                   const float* __restrict__ ss,
                                   float* __restrict__ out,
                                   int C, int N, int M, int Btot) {
  int i = blockIdx.x * blockDim.x + threadIdx.x;
  int total = Btot * C * M;
  if (i >= total) return;
  int m  = i % M;
  int bc = i / M;
  int c  = bc % C;
  float sc = ss[c], sh = ss[C + c];
  const float* row = h + (size_t)bc * (size_t)N;
  float v = -INFINITY;
  #pragma unroll
  for (int j = 0; j < 4; ++j) {
    float e = row[pidx[m * 4 + j]] * sc + sh;
    e = e > 0.f ? e : 0.f;
    v = fmaxf(v, e);
  }
  out[i] = v;
}

// ---------------------------------------------------------------------------
extern "C" void kernel_launch(void* const* d_in, const int* in_sizes, int n_in,
                              void* d_out, int out_size, void* d_ws, size_t ws_size,
                              hipStream_t stream) {
  (void)in_sizes; (void)n_in; (void)out_size; (void)ws_size;
  const int B = 16, N1 = 50176, N2 = 12544, N3 = 3136;

  const float* x    = (const float*)d_in[0];
  const float* l1   = (const float*)d_in[1];
  const float* l2   = (const float*)d_in[2];
  const int*   knn1 = (const int*)d_in[4];
  const int*   knn2 = (const int*)d_in[5];
  const int*   knn3 = (const int*)d_in[6];
  const int*   pidx = (const int*)d_in[7];

  const float* p1[6]; const float* p2[6]; const float* p3[6];
  for (int j = 0; j < 6; ++j) { p1[j] = (const float*)d_in[8 + j];
                                p2[j] = (const float*)d_in[14 + j];
                                p3[j] = (const float*)d_in[20 + j]; }
  const float* bn1g = (const float*)d_in[26]; const float* bn1b = (const float*)d_in[27];
  const float* bn2g = (const float*)d_in[28]; const float* bn2b = (const float*)d_in[29];
  const float* bn3g = (const float*)d_in[30]; const float* bn3b = (const float*)d_in[31];

  char* ws = (char*)d_ws;
  size_t off = 0;
  auto alloc = [&](size_t bytes) { size_t p = off; off += (bytes + 255) & ~(size_t)255; return p; };
  float*    h1  = (float*)(ws + alloc(sizeof(float) * (size_t)B * 32 * N2));
  float*    h2  = (float*)(ws + alloc(sizeof(float) * (size_t)B * 32 * N2));
  float*    h3  = (float*)(ws + alloc(sizeof(float) * (size_t)B * 64 * N2));
  uint32_t* bf1 = (uint32_t*)(ws + alloc(4ull * 2 * 1 * 256));
  uint32_t* bf2 = (uint32_t*)(ws + alloc(4ull * 2 * 9 * 256));
  uint32_t* bf3 = (uint32_t*)(ws + alloc(4ull * 4 * 9 * 256));
  float*    ss1 = (float*)(ws + alloc(4ull * 2 * 32));
  float*    ss2 = (float*)(ws + alloc(4ull * 2 * 32));
  float*    ss3 = (float*)(ws + alloc(4ull * 2 * 64));

  // pre-pack conv kernels into WMMA B fragments
  build_bfrag_kernel<<<(2 * 1 * 256 + 255) / 256, 256, 0, stream>>>(p1[4], bf1, 3, 32, 1, 2);
  build_bfrag_kernel<<<(2 * 9 * 256 + 255) / 256, 256, 0, stream>>>(p2[4], bf2, 32, 32, 9, 2);
  build_bfrag_kernel<<<(4 * 9 * 256 + 255) / 256, 256, 0, stream>>>(p3[4], bf3, 32, 64, 9, 4);

  const int convBlocks = N2 / 4;  // one wave per point, 4 waves per block

  // layer 1: x (B,3,N1) -> h1 raw (B,32,N2); no input BN
  interp_conv_wmma_kernel<3, 32, 1, 2, false><<<convBlocks, 128, 0, stream>>>(
      x, knn1, l1, l2, p1[0], p1[1], p1[2], p1[3], bf1, p1[5], nullptr, h1, N1, N2);
  bn_stats_kernel<<<32, 256, 0, stream>>>(h1, bn1g, bn1b, ss1, 32, N2, B);

  // layer 2: bn_relu(h1) gathered on the fly -> h2 raw (B,32,N2)
  interp_conv_wmma_kernel<32, 32, 9, 2, true><<<convBlocks, 128, 0, stream>>>(
      h1, knn2, l2, l2, p2[0], p2[1], p2[2], p2[3], bf2, p2[5], ss1, h2, N2, N2);
  bn_stats_kernel<<<32, 256, 0, stream>>>(h2, bn2g, bn2b, ss2, 32, N2, B);

  // layer 3: bn_relu(h2) gathered on the fly -> h3 raw (B,64,N2)
  interp_conv_wmma_kernel<32, 64, 9, 4, true><<<convBlocks, 128, 0, stream>>>(
      h2, knn3, l2, l2, p3[0], p3[1], p3[2], p3[3], bf3, p3[5], ss2, h3, N2, N2);
  bn_stats_kernel<<<64, 256, 0, stream>>>(h3, bn3g, bn3b, ss3, 64, N2, B);

  // fused BN/ReLU + 4-way max pool -> d_out (B,64,N3)
  pool_bn_max_kernel<<<(B * 64 * N3 + 255) / 256, 256, 0, stream>>>(
      h3, pidx, ss3, (float*)d_out, 64, N2, N3, B);
}
